// MultiHeadSelfAttention_16973710754359
// MI455X (gfx1250) — matrix operand order repaired and checked by reading, not compiled, not run on hardware
//
#include <hip/hip_runtime.h>

#define DIM     512
#define NHEADS  8
#define HD      64
#define SEQ     2048
#define BATCH   4
#define ROWS    (BATCH * SEQ)   // 8192
#define SCALE_F 0.125f          // HEAD_DIM ** -0.5

typedef __attribute__((ext_vector_type(16))) __bf16 v16bf;
typedef __attribute__((ext_vector_type(8)))  float  v8f;
typedef __attribute__((ext_vector_type(8)))  __bf16 bf16x8;
typedef __attribute__((ext_vector_type(4)))  float  f32x4;
typedef __attribute__((ext_vector_type(4)))  unsigned int u32x4;
typedef __attribute__((ext_vector_type(8)))  int    i32x8;
typedef __attribute__((ext_vector_type(4)))  int    i32x4;

// ---------------------------------------------------------------------------
// Kernel 1: QKV projection.  C[8192,1536] = X[8192,512] @ Wqkv[512,1536]
// One 16x16 output tile per wave, K-loop of 32, bf16 WMMA with f32 accum.
// ---------------------------------------------------------------------------
__global__ __launch_bounds__(256) void qkv_kernel(
    const float* __restrict__ X, const float* __restrict__ W,
    __bf16* __restrict__ qbf, __bf16* __restrict__ kbf,
    __bf16* __restrict__ vbf) {
    const int lane = threadIdx.x & 31;
    const int wave = threadIdx.x >> 5;
    const int l16  = lane & 15;
    const int hl   = lane >> 4;          // lane half (0/1)

    const int tile = blockIdx.x * 8 + wave;
    const int TN   = (3 * DIM) / 16;     // 96 col tiles
    const int tn   = tile % TN;
    const int tm   = tile / TN;
    const int m0   = tm * 16;
    const int n0   = tn * 16;

    v8f acc = {0.f, 0.f, 0.f, 0.f, 0.f, 0.f, 0.f, 0.f};

    const float* Xrow = X + (size_t)(m0 + l16) * DIM;
    for (int k0 = 0; k0 < DIM; k0 += 32) {
        // Prefetch next k-block of X (global_prefetch_b8)
        if (k0 + 32 < DIM)
            __builtin_prefetch((const void*)(Xrow + k0 + 32 + (hl << 3)), 0, 0);
        v16bf a, b;
        __bf16* ap = (__bf16*)&a;
        __bf16* bp = (__bf16*)&b;
        // A fragment: e<8 -> k = k0 + 8*hl + e ; e>=8 -> k = k0 + 16 + 8*hl + (e-8)
        const f32x4 x0 = *(const f32x4*)(Xrow + k0 + (hl << 3));
        const f32x4 x1 = *(const f32x4*)(Xrow + k0 + (hl << 3) + 4);
        const f32x4 x2 = *(const f32x4*)(Xrow + k0 + 16 + (hl << 3));
        const f32x4 x3 = *(const f32x4*)(Xrow + k0 + 16 + (hl << 3) + 4);
#pragma unroll
        for (int j = 0; j < 4; ++j) {
            ap[j]      = (__bf16)x0[j];
            ap[4 + j]  = (__bf16)x1[j];
            ap[8 + j]  = (__bf16)x2[j];
            ap[12 + j] = (__bf16)x3[j];
        }
        // B fragment: e<8 -> k = k0 + 8*hl + e ; e>=8 -> k = k0 + 16 + 8*hl + (e-8), col = n0 + l16 (coalesced across lanes)
#pragma unroll
        for (int e = 0; e < 16; ++e) {
            const int kb = k0 + (hl << 3) + (e & 7) + ((e >> 3) << 4);
            bp[e] = (__bf16)W[kb * (3 * DIM) + n0 + l16];
        }
        acc = __builtin_amdgcn_wmma_f32_16x16x32_bf16(
            false, a, false, b, (short)0, acc, false, false);
    }

    // col -> (s in {q,k,v}, head h, d); constant across the 16-wide tile
    const int col = n0 + l16;
    const int s   = col >> 9;            // / 512
    const int h   = (col >> 6) & 7;
    const int d   = col & 63;
    __bf16* dst = (s == 0) ? qbf : (s == 1) ? kbf : vbf;
    const float sc = (s == 0) ? SCALE_F : 1.0f;
#pragma unroll
    for (int v = 0; v < 8; ++v) {
        const int row = m0 + v + (hl << 3);
        const int b_  = row >> 11;       // / 2048
        const int n   = row & 2047;
        dst[((size_t)((b_ * NHEADS + h) * SEQ + n) << 6) + d] = (__bf16)(acc[v] * sc);
    }
}

// ---------------------------------------------------------------------------
// Kernel 2: flash attention per (b,h).  Wave owns 16 query rows x hd=64; all 8
// waves of a block share (b,h) and the same key-tile sequence.  The K tile
// (32 keys x 64 d bf16 = one contiguous 4 KB block) is DMA'd into LDS by the
// Tensor Data Mover (wave 0 builds the D# descriptor, issues
// tensor_load_to_lds, drains TENSORcnt); the V tile is staged transposed
// [d][key] by all threads so both B-fragment reads are contiguous
// ds_load_b128.  S = Q K^T (4 WMMAs), online softmax via shfl reductions in
// the 16-lane halves, P bounced through per-wave LDS (s_wait_dscnt for the
// cross-lane RAW), then O += P V (4 WMMAs).
// ---------------------------------------------------------------------------
__global__ __launch_bounds__(256) void attn_kernel(
    const __bf16* __restrict__ qbf, const __bf16* __restrict__ kbf,
    const __bf16* __restrict__ vbf, __bf16* __restrict__ obf) {
    __shared__ __bf16 kst[32 * 64];      // K tile  [key][d]      4 KB (TDM dest)
    __shared__ __bf16 vst[64 * 32];      // V tile  [d][key]      4 KB
    __shared__ __bf16 pst[8 * 16 * 32];  // P tile per wave       8 KB

    const int tid  = threadIdx.x;
    const int lane = tid & 31;
    const int wave = tid >> 5;
    const int l16  = lane & 15;
    const int hl   = lane >> 4;

    const int bh   = blockIdx.x >> 4;    // 0..31
    const int qblk = blockIdx.x & 15;
    const int q0   = qblk * 128 + wave * 16;

    const __bf16* Q = qbf + (size_t)bh * SEQ * HD;
    const __bf16* K = kbf + (size_t)bh * SEQ * HD;
    const __bf16* V = vbf + (size_t)bh * SEQ * HD;
    __bf16* pw = pst + wave * (16 * 32);

    // Loop-invariant Q A-fragments (d=0..31 and d=32..63), b128 loads
    v16bf aq0, aq1;
    {
        const __bf16* qrow = Q + (size_t)(q0 + l16) * HD;
        bf16x8* a0 = (bf16x8*)&aq0;
        bf16x8* a1 = (bf16x8*)&aq1;
        a0[0] = *(const bf16x8*)(qrow + (hl << 4));          // e0..7 : d = 16*hl+e
        a0[1] = *(const bf16x8*)(qrow + 8 + (hl << 4));      // e8..15: d = 16*hl+e
        a1[0] = *(const bf16x8*)(qrow + 32 + (hl << 4));
        a1[1] = *(const bf16x8*)(qrow + 40 + (hl << 4));
    }

    v8f o0 = {0.f,0.f,0.f,0.f,0.f,0.f,0.f,0.f};
    v8f o1 = o0, o2 = o0, o3 = o0;
    float mr[8], sr[8];
#pragma unroll
    for (int v = 0; v < 8; ++v) { mr[v] = -1e30f; sr[v] = 0.f; }

    // TDM D# pieces that are loop-invariant (group 1, and LDS dest offset)
    const unsigned lds_kst = (unsigned)(unsigned long long)(uintptr_t)kst;
    i32x8 tdm_g1;
    tdm_g1[0] = (int)(1u << 16);              // workgroup_mask=0, data_size=1 (2B)
    tdm_g1[1] = (int)((unsigned)HD << 16);    // tensor_dim0[15:0]=64 @ bits 63:48
    tdm_g1[2] = (int)((unsigned)SEQ << 16);   // tensor_dim1[15:0]=2048 @ bits 95:80
    tdm_g1[3] = (int)((unsigned)HD << 16);    // tile_dim0=64 @ bits 127:112
    tdm_g1[4] = 32;                           // tile_dim1=32 rows, tile_dim2=0
    tdm_g1[5] = HD;                           // tensor_dim0_stride=64 (low bits)
    tdm_g1[6] = 0;
    tdm_g1[7] = 0;
    const i32x4 tdm_z4 = {0, 0, 0, 0};
    const i32x8 tdm_z8 = {0, 0, 0, 0, 0, 0, 0, 0};

    // Cooperative V staging: 256 threads x 8 bf16 = one 32x64 tile
    const int skey = tid >> 3;           // 0..31
    const int sdof = (tid & 7) << 3;     // 0,8,...,56

    for (int kt = 0; kt < SEQ; kt += 32) {
        // ---- K tile via Tensor Data Mover (one descriptor per block) -------
        if (wave == 0) {
            const unsigned long long ga =
                (unsigned long long)(uintptr_t)(K + (size_t)kt * HD);
            u32x4 g0;
            g0[0] = 1u;                                   // count=1, user mode
            g0[1] = lds_kst;                              // lds_addr (bytes)
            g0[2] = (unsigned)(ga & 0xFFFFFFFFull);       // global_addr[31:0]
            g0[3] = (unsigned)((ga >> 32) & 0x1FFFFFFull) // global_addr[56:32]
                    | 0x80000000u;                        // type=2 ("image")
            __builtin_amdgcn_tensor_load_to_lds(g0, tdm_g1, tdm_z4, tdm_z4,
                                                tdm_z8, 0);
        }
        // ---- stage V^T tile into LDS (needs transpose; TDM can't) ----------
        const bf16x8 vv = *(const bf16x8*)(V + (size_t)(kt + skey) * HD + sdof);
#pragma unroll
        for (int j = 0; j < 8; ++j)
            vst[(sdof + j) * 32 + skey] = vv[j];
        if (wave == 0)
            __builtin_amdgcn_s_wait_tensorcnt(0);
        __syncthreads();

        // ---- S = Q K^T : B-fragments from kst (contiguous over e) ----------
        v16bf bk00, bk01, bk10, bk11;
        {
            const __bf16* kr0 = kst + l16 * 64 + (hl << 4);        // keys 0..15
            const __bf16* kr1 = kst + (16 + l16) * 64 + (hl << 4); // keys 16..31
            ((bf16x8*)&bk00)[0] = *(const bf16x8*)(kr0);
            ((bf16x8*)&bk00)[1] = *(const bf16x8*)(kr0 + 8);
            ((bf16x8*)&bk01)[0] = *(const bf16x8*)(kr0 + 32);
            ((bf16x8*)&bk01)[1] = *(const bf16x8*)(kr0 + 40);
            ((bf16x8*)&bk10)[0] = *(const bf16x8*)(kr1);
            ((bf16x8*)&bk10)[1] = *(const bf16x8*)(kr1 + 8);
            ((bf16x8*)&bk11)[0] = *(const bf16x8*)(kr1 + 32);
            ((bf16x8*)&bk11)[1] = *(const bf16x8*)(kr1 + 40);
        }
        v8f s0 = {0.f,0.f,0.f,0.f,0.f,0.f,0.f,0.f};
        v8f s1 = s0;
        s0 = __builtin_amdgcn_wmma_f32_16x16x32_bf16(false, aq0, false, bk00, (short)0, s0, false, false);
        s0 = __builtin_amdgcn_wmma_f32_16x16x32_bf16(false, aq1, false, bk01, (short)0, s0, false, false);
        s1 = __builtin_amdgcn_wmma_f32_16x16x32_bf16(false, aq0, false, bk10, (short)0, s1, false, false);
        s1 = __builtin_amdgcn_wmma_f32_16x16x32_bf16(false, aq1, false, bk11, (short)0, s1, false, false);

        // ---- online softmax; row r = v + 8*hl lives on one 16-lane half ----
#pragma unroll
        for (int v = 0; v < 8; ++v) {
            float t = fmaxf(s0[v], s1[v]);
            t = fmaxf(t, __shfl_xor(t, 1, 32));
            t = fmaxf(t, __shfl_xor(t, 2, 32));
            t = fmaxf(t, __shfl_xor(t, 4, 32));
            t = fmaxf(t, __shfl_xor(t, 8, 32));
            const float nm = fmaxf(mr[v], t);
            const float f  = __expf(mr[v] - nm);
            const float p0 = __expf(s0[v] - nm);
            const float p1 = __expf(s1[v] - nm);
            float ps = p0 + p1;
            ps += __shfl_xor(ps, 1, 32);
            ps += __shfl_xor(ps, 2, 32);
            ps += __shfl_xor(ps, 4, 32);
            ps += __shfl_xor(ps, 8, 32);
            sr[v] = sr[v] * f + ps;
            mr[v] = nm;
            o0[v] *= f; o1[v] *= f; o2[v] *= f; o3[v] *= f;
            const int row = v + (hl << 3);
            pw[row * 32 + l16]      = (__bf16)p0;
            pw[row * 32 + 16 + l16] = (__bf16)p1;
        }
        // Cross-lane LDS RAW within the wave: drain DS counter explicitly
        asm volatile("s_wait_dscnt 0" ::: "memory");

        // ---- P as A-fragment (two contiguous b128 reads) -------------------
        v16bf apf;
        {
            const __bf16* pr = pw + l16 * 32 + (hl << 4);
            ((bf16x8*)&apf)[0] = *(const bf16x8*)(pr);        // e0..7 : k=16*hl+e
            ((bf16x8*)&apf)[1] = *(const bf16x8*)(pr + 8);    // e8..15: k=16*hl+e
        }
        // ---- V B-fragments from transposed vst (contiguous over e) ---------
        v16bf bv0, bv1, bv2, bv3;
        {
            const __bf16* v0 = vst + l16 * 32 + (hl << 4);
            const __bf16* v1 = vst + (16 + l16) * 32 + (hl << 4);
            const __bf16* v2 = vst + (32 + l16) * 32 + (hl << 4);
            const __bf16* v3 = vst + (48 + l16) * 32 + (hl << 4);
            ((bf16x8*)&bv0)[0] = *(const bf16x8*)(v0);
            ((bf16x8*)&bv0)[1] = *(const bf16x8*)(v0 + 8);
            ((bf16x8*)&bv1)[0] = *(const bf16x8*)(v1);
            ((bf16x8*)&bv1)[1] = *(const bf16x8*)(v1 + 8);
            ((bf16x8*)&bv2)[0] = *(const bf16x8*)(v2);
            ((bf16x8*)&bv2)[1] = *(const bf16x8*)(v2 + 8);
            ((bf16x8*)&bv3)[0] = *(const bf16x8*)(v3);
            ((bf16x8*)&bv3)[1] = *(const bf16x8*)(v3 + 8);
        }
        o0 = __builtin_amdgcn_wmma_f32_16x16x32_bf16(false, apf, false, bv0, (short)0, o0, false, false);
        o1 = __builtin_amdgcn_wmma_f32_16x16x32_bf16(false, apf, false, bv1, (short)0, o1, false, false);
        o2 = __builtin_amdgcn_wmma_f32_16x16x32_bf16(false, apf, false, bv2, (short)0, o2, false, false);
        o3 = __builtin_amdgcn_wmma_f32_16x16x32_bf16(false, apf, false, bv3, (short)0, o3, false, false);

        __syncthreads();   // protect kst/vst before next iteration's staging
    }

    // Epilogue: divide by softmax denominator, store bf16 to [B*N, DIM]
    const int b_ = bh >> 3;
    const int h  = bh & 7;
#pragma unroll
    for (int v = 0; v < 8; ++v) {
        const float inv = 1.0f / sr[v];
        const int n  = q0 + v + (hl << 3);
        __bf16* r = obf + (size_t)(b_ * SEQ + n) * DIM + h * HD + l16;
        r[0]  = (__bf16)(o0[v] * inv);
        r[16] = (__bf16)(o1[v] * inv);
        r[32] = (__bf16)(o2[v] * inv);
        r[48] = (__bf16)(o3[v] * inv);
    }
}

// ---------------------------------------------------------------------------
// Kernel 3: output projection.  out[8192,512] = O[8192,512] @ Wout + b_out
// ---------------------------------------------------------------------------
__global__ __launch_bounds__(256) void proj_kernel(
    const __bf16* __restrict__ O, const float* __restrict__ W,
    const float* __restrict__ bias, float* __restrict__ out) {
    const int lane = threadIdx.x & 31;
    const int wave = threadIdx.x >> 5;
    const int l16  = lane & 15;
    const int hl   = lane >> 4;

    const int tile = blockIdx.x * 8 + wave;
    const int tn   = tile & 31;          // DIM/16 = 32 col tiles
    const int tm   = tile >> 5;
    const int m0   = tm * 16;
    const int n0   = tn * 16;

    v8f acc = {0.f,0.f,0.f,0.f,0.f,0.f,0.f,0.f};

    const __bf16* Orow = O + (size_t)(m0 + l16) * DIM;
    for (int k0 = 0; k0 < DIM; k0 += 32) {
        v16bf a, b;
        // A fragment: two contiguous b128 loads of bf16
        ((bf16x8*)&a)[0] = *(const bf16x8*)(Orow + k0 + (hl << 3));
        ((bf16x8*)&a)[1] = *(const bf16x8*)(Orow + k0 + 16 + (hl << 3));
        __bf16* bp = (__bf16*)&b;
#pragma unroll
        for (int e = 0; e < 16; ++e) {
            const int kb = k0 + (hl << 3) + (e & 7) + ((e >> 3) << 4);
            bp[e] = (__bf16)W[kb * DIM + n0 + l16];
        }
        acc = __builtin_amdgcn_wmma_f32_16x16x32_bf16(
            false, a, false, b, (short)0, acc, false, false);
    }

    const float bv = bias[n0 + l16];
#pragma unroll
    for (int v = 0; v < 8; ++v) {
        const int row = m0 + v + (hl << 3);
        out[(size_t)row * DIM + n0 + l16] = acc[v] + bv;
    }
}

// ---------------------------------------------------------------------------
extern "C" void kernel_launch(void* const* d_in, const int* in_sizes, int n_in,
                              void* d_out, int out_size, void* d_ws, size_t ws_size,
                              hipStream_t stream) {
    const float* x     = (const float*)d_in[0];   // [4,2048,512]
    const float* w_qkv = (const float*)d_in[1];   // [512,1536]
    const float* w_out = (const float*)d_in[2];   // [512,512]
    const float* b_out = (const float*)d_in[3];   // [512]
    float* out = (float*)d_out;                   // [4,2048,512]

    const size_t HBUF = (size_t)BATCH * NHEADS * SEQ * HD;  // 4,194,304 elems
    __bf16* qbf = (__bf16*)d_ws;
    __bf16* kbf = qbf + HBUF;
    __bf16* vbf = kbf + HBUF;
    __bf16* obf = vbf + HBUF;

    // 8192x1536 in 16x16 tiles = 49152 wave-tiles; 8 waves per 256-thread block
    qkv_kernel<<<6144, 256, 0, stream>>>(x, w_qkv, qbf, kbf, vbf);
    // 32 (b,h) pairs x 16 query blocks of 128 rows
    attn_kernel<<<512, 256, 0, stream>>>(qbf, kbf, vbf, obf);
    // 8192x512 in 16x16 tiles = 16384 wave-tiles
    proj_kernel<<<2048, 256, 0, stream>>>(obf, w_out, b_out, out);
}
